// EncoderAttention_68444598829075
// MI455X (gfx1250) — compile-verified, matrix operand order checked by reading, not run on hardware
//
#include <hip/hip_runtime.h>

// Problem constants (from the reference setup)
#define B_ 2
#define S_ 2048
#define E_ 1024
#define H_ 16
#define D_ 64

// GEMM block tiling: 8 waves (2M x 4N), wave tile 32x64, K-step 32
#define TM 64
#define TN 256
#define TK 32

typedef __bf16 bf16;
typedef __attribute__((ext_vector_type(16))) __bf16 v16bf;
typedef __attribute__((ext_vector_type(8)))  float  v8f;

#if defined(__gfx1250__) && __has_builtin(__builtin_amdgcn_global_load_async_to_lds_b128)
#define USE_ASYNC_LDS 1
#else
#define USE_ASYNC_LDS 0
#endif

// Pointer types the async builtin actually wants (from hipcc's diagnostic):
// int __attribute__((vector_size(16))) * in addrspace(1) / addrspace(3).
typedef int vsi4 __attribute__((vector_size(16)));
typedef __attribute__((address_space(1))) vsi4 gvsi4;
typedef __attribute__((address_space(3))) vsi4 svsi4;

// ---------------------------------------------------------------------------
// Copy primitives: async global->LDS (ASYNCcnt path) with reg-staged fallback
// ---------------------------------------------------------------------------
__device__ __forceinline__ void cp_b128(const bf16* __restrict__ g, bf16* l) {
#if USE_ASYNC_LDS
  __builtin_amdgcn_global_load_async_to_lds_b128((gvsi4*)(void*)g,
                                                 (svsi4*)(void*)l, 0, 0);
#else
  *(uint4*)l = *(const uint4*)g;
#endif
}

__device__ __forceinline__ void cp_commit() {
#if USE_ASYNC_LDS
#if __has_builtin(__builtin_amdgcn_s_wait_asynccnt)
  __builtin_amdgcn_s_wait_asynccnt(0);
#else
  asm volatile("s_wait_asynccnt 0" ::: "memory");
#endif
#endif
}

// ---------------------------------------------------------------------------
// WMMA helpers (CDNA5 wave32, v_wmma_f32_16x16x32_bf16)
// A-frag (16x32 bf16): lane r(0..15)/half h holds row M=r, two K-contiguous
// 16-byte chunks at (r*ld + 8h) and (+16). The B-frag of X*W^T (and Q*K^T)
// loads with the identical pattern from row-major W / K rows.
// ---------------------------------------------------------------------------
__device__ __forceinline__ v16bf load_frag_b128(const bf16* __restrict__ src, int ld) {
  const int l = threadIdx.x & 31;
  const int r = l & 15;
  const int h = l >> 4;
  const bf16* p = src + r * ld + h * 8;
  union { v16bf v; uint4 u[2]; } u;
  u.u[0] = *(const uint4*)(p);
  u.u[1] = *(const uint4*)(p + 16);
  return u.v;
}

__device__ __forceinline__ v8f wmma_bf16(v16bf a, v16bf b, v8f c) {
  return __builtin_amdgcn_wmma_f32_16x16x32_bf16(false, a, false, b, (short)0, c,
                                                 false, false);
}

__device__ __forceinline__ v8f zero8() {
  v8f z;
#pragma unroll
  for (int i = 0; i < 8; ++i) z[i] = 0.0f;
  return z;
}

// ---------------------------------------------------------------------------
// fp32 -> bf16 cast
// ---------------------------------------------------------------------------
__global__ __launch_bounds__(256) void cast_f32_bf16(const float* __restrict__ in,
                                                     bf16* __restrict__ out, int n) {
  int i = blockIdx.x * 256 + threadIdx.x;
  if (i < n) out[i] = (bf16)in[i];
}

// ---------------------------------------------------------------------------
// Cooperative tile copy: A tile 64x32 (1 b128/thread), B tile 256x32
// (4 b128/thread). ld is the row stride (K dim) of both global matrices.
// ---------------------------------------------------------------------------
__device__ __forceinline__ void copy_tiles(const bf16* __restrict__ gA,
                                           const bf16* __restrict__ gB, int ld,
                                           bf16* lA, bf16* lB) {
  const int t = threadIdx.x;
  const int row = t >> 2;
  const int c = (t & 3) * 8;
  cp_b128(gA + (size_t)row * ld + c, lA + row * TK + c);
#pragma unroll
  for (int i = 0; i < 4; ++i)
    cp_b128(gB + (size_t)(row + 64 * i) * ld + c, lB + (row + 64 * i) * TK + c);
}

// ---------------------------------------------------------------------------
// Double-buffered LDS-staged GEMM core: acc[2][4] += A(64xK) * B(256xK)^T
// for this wave's 32x64 sub-tile. 8 WMMA per K-step from ds_load_b128 frags,
// while the next tiles stream in on the async path.
// ---------------------------------------------------------------------------
__device__ __forceinline__ void gemm_core(const bf16* __restrict__ gA,
                                          const bf16* __restrict__ gB, int Kdim,
                                          bf16* sA, bf16* sB, v8f acc[2][4]) {
  const int w  = threadIdx.x >> 5;
  const int wm = w >> 2, wn = w & 3;

  copy_tiles(gA, gB, Kdim, sA, sB);
  cp_commit();
  __syncthreads();

  int buf = 0;
  for (int k0 = 0; k0 < Kdim; k0 += TK) {
    if (k0 + TK < Kdim)
      copy_tiles(gA + k0 + TK, gB + k0 + TK, Kdim,
                 sA + (buf ^ 1) * (TM * TK), sB + (buf ^ 1) * (TN * TK));

    const bf16* lA = sA + buf * (TM * TK) + (wm * 32) * TK;
    const bf16* lB = sB + buf * (TN * TK) + (wn * 64) * TK;
    v16bf af[2], bfr[4];
#pragma unroll
    for (int i = 0; i < 2; ++i) af[i] = load_frag_b128(lA + 16 * i * TK, TK);
#pragma unroll
    for (int j = 0; j < 4; ++j) bfr[j] = load_frag_b128(lB + 16 * j * TK, TK);
#pragma unroll
    for (int i = 0; i < 2; ++i)
#pragma unroll
      for (int j = 0; j < 4; ++j) acc[i][j] = wmma_bf16(af[i], bfr[j], acc[i][j]);

    cp_commit();
    __syncthreads();
    buf ^= 1;
  }
}

// ---------------------------------------------------------------------------
// QKV projection: qkv = X[4096,1024] * Win^T[1024,3072] + b_in.
// Epilogue scatters Q,K at natural [B,S,E] layout (== the reference's flat
// [B,H,S,D] view) and V transposed to [B,H,D,S] for contiguous P@V frags.
// ---------------------------------------------------------------------------
__global__ __launch_bounds__(256) void qkv_kernel(const bf16* __restrict__ X,
                                                  const bf16* __restrict__ W,
                                                  const float* __restrict__ bias,
                                                  bf16* __restrict__ Q,
                                                  bf16* __restrict__ Kt,
                                                  bf16* __restrict__ Vt) {
  __shared__ bf16 sA[2 * TM * TK];
  __shared__ bf16 sB[2 * TN * TK];
  const int w  = threadIdx.x >> 5;
  const int wm = w >> 2, wn = w & 3;
  const int mB = blockIdx.x * TM;
  const int nB = blockIdx.y * TN;
  const int l = threadIdx.x & 31, r = l & 15, hf = l >> 4;

  v8f acc[2][4];
#pragma unroll
  for (int i = 0; i < 2; ++i)
#pragma unroll
    for (int j = 0; j < 4; ++j) acc[i][j] = zero8();

  gemm_core(X + (size_t)mB * E_, W + (size_t)nB * E_, E_, sA, sB, acc);

  const int m0 = mB + wm * 32;
  const int n0 = nB + wn * 64;
#pragma unroll
  for (int i = 0; i < 2; ++i)
#pragma unroll
    for (int j = 0; j < 4; ++j) {
      const int nb = n0 + 16 * j + r;
      const float bv = bias[nb];
#pragma unroll
      for (int t = 0; t < 8; ++t) {
        const int m = m0 + 16 * i + t + 8 * hf;  // C-layout row
        const bf16 o = (bf16)(acc[i][j][t] + bv);
        const int b = m >> 11;       // m / S_
        const int s = m & (S_ - 1);  // m % S_
        if (nb < E_) {
          Q[((size_t)b * S_ + s) * E_ + nb] = o;
        } else if (nb < 2 * E_) {
          Kt[((size_t)b * S_ + s) * E_ + (nb - E_)] = o;
        } else {
          const int e  = nb - 2 * E_;
          const int hh = s >> 7;                       // reference's flat head id
          const int sh = ((s & 127) << 4) | (e >> 6);  // head-internal seq
          const int dd = e & 63;
          Vt[(((size_t)b * H_ + hh) * D_ + dd) * S_ + sh] = o;
        }
      }
    }
}

// ---------------------------------------------------------------------------
// Flash attention, one wave per 16-query tile of one (b,h); online softmax in
// fp32, P restaged C-layout -> A-layout through per-wave LDS (ld=40 keeps the
// fragment chunks 16B aligned -> ds_load_b128), fenced by s_wait_dscnt.
// ---------------------------------------------------------------------------
__global__ __launch_bounds__(256) void attn_kernel(const bf16* __restrict__ Q,
                                                   const bf16* __restrict__ Kt,
                                                   const bf16* __restrict__ Vt,
                                                   bf16* __restrict__ Ctx,
                                                   const int* __restrict__ causal_p) {
  __shared__ bf16 pbuf[8 * 16 * 40];
  const int w = threadIdx.x >> 5;
  const int l = threadIdx.x & 31, r = l & 15, hf = l >> 4;
  const int bh   = blockIdx.x >> 4;  // (b*H + h)
  const int qblk = blockIdx.x & 15;
  const int q0   = qblk * 128 + w * 16;
  const int causal = *causal_p;

  const bf16* qb = Q  + (size_t)bh * S_ * D_ + (size_t)q0 * D_;
  const bf16* kb = Kt + (size_t)bh * S_ * D_;
  const bf16* vb = Vt + (size_t)bh * D_ * S_;
  bf16* pb = pbuf + w * 16 * 40;

  v16bf aq[2];
  aq[0] = load_frag_b128(qb + 0, D_);
  aq[1] = load_frag_b128(qb + 32, D_);

  v8f ctx[4];
#pragma unroll
  for (int ni = 0; ni < 4; ++ni) ctx[ni] = zero8();
  float mrow[8], srow[8];
#pragma unroll
  for (int j = 0; j < 8; ++j) { mrow[j] = -1e30f; srow[j] = 0.0f; }

  const float scale = 0.125f;  // 1/sqrt(D)
  const int nblk = causal ? (q0 / 32 + 1) : (S_ / 32);

  for (int blk = 0; blk < nblk; ++blk) {
    const int k0 = blk * 32;
    v16bf bk[2][2];
#pragma unroll
    for (int t = 0; t < 2; ++t)
#pragma unroll
      for (int ks = 0; ks < 2; ++ks)
        bk[t][ks] = load_frag_b128(kb + (size_t)(k0 + 16 * t) * D_ + 32 * ks, D_);
    if (blk + 1 < nblk) {
      __builtin_prefetch(kb + (size_t)(k0 + 32) * D_, 0, 1);
      __builtin_prefetch(vb + (size_t)0 * S_ + k0 + 32, 0, 1);
    }

    v8f s0 = zero8(), s1 = zero8();
    s0 = wmma_bf16(aq[0], bk[0][0], s0);
    s0 = wmma_bf16(aq[1], bk[0][1], s0);
    s1 = wmma_bf16(aq[0], bk[1][0], s1);
    s1 = wmma_bf16(aq[1], bk[1][1], s1);

    float p0[8], p1[8], corr[8];
#pragma unroll
    for (int j = 0; j < 8; ++j) {
      float t0 = s0[j] * scale;
      float t1 = s1[j] * scale;
      if (causal) {
        const int qi = q0 + j + 8 * hf;
        if (k0 + r > qi)      t0 = -1e30f;
        if (k0 + 16 + r > qi) t1 = -1e30f;
      }
      float ml = fmaxf(t0, t1);
#pragma unroll
      for (int off = 8; off > 0; off >>= 1)
        ml = fmaxf(ml, __shfl_xor(ml, off, 32));  // reduce within 16-lane half
      const float mnew = fmaxf(mrow[j], ml);
      corr[j] = __expf(mrow[j] - mnew);
      t0 = __expf(t0 - mnew);
      t1 = __expf(t1 - mnew);
      float ssum = t0 + t1;
#pragma unroll
      for (int off = 8; off > 0; off >>= 1)
        ssum += __shfl_xor(ssum, off, 32);
      srow[j] = srow[j] * corr[j] + ssum;
      mrow[j] = mnew;
      p0[j] = t0;
      p1[j] = t1;
    }

    // C-layout -> A-layout restage (intra-wave; LDS is in-order per wave,
    // s_wait_dscnt 0 orders the dependent load and fences the compiler).
#pragma unroll
    for (int j = 0; j < 8; ++j) {
      pb[(j + 8 * hf) * 40 + r]      = (bf16)p0[j];
      pb[(j + 8 * hf) * 40 + 16 + r] = (bf16)p1[j];
    }
    asm volatile("s_wait_dscnt 0" ::: "memory");
    const v16bf pf = load_frag_b128(pb, 40);

#pragma unroll
    for (int ni = 0; ni < 4; ++ni) {
#pragma unroll
      for (int j = 0; j < 8; ++j) ctx[ni][j] *= corr[j];
      const v16bf bv = load_frag_b128(vb + (size_t)(16 * ni) * S_ + k0, S_);
      ctx[ni] = wmma_bf16(pf, bv, ctx[ni]);
    }
  }

  // Reference output transpose (0,2,1,3): ctx[b,h,s,d] -> Ctx[b][s][h*64+d]
  const int hh = bh & (H_ - 1);
  const int bb = bh >> 4;
#pragma unroll
  for (int ni = 0; ni < 4; ++ni)
#pragma unroll
    for (int j = 0; j < 8; ++j) {
      const int qi = q0 + j + 8 * hf;
      const int col = hh * D_ + 16 * ni + r;
      Ctx[((size_t)bb * S_ + qi) * E_ + col] = (bf16)(ctx[ni][j] / srow[j]);
    }
}

// ---------------------------------------------------------------------------
// Output projection: out = Ctx[4096,1024] * Wout^T[1024,1024] + b_out (fp32)
// ---------------------------------------------------------------------------
__global__ __launch_bounds__(256) void out_kernel(const bf16* __restrict__ Ctx,
                                                  const bf16* __restrict__ W,
                                                  const float* __restrict__ bias,
                                                  float* __restrict__ out) {
  __shared__ bf16 sA[2 * TM * TK];
  __shared__ bf16 sB[2 * TN * TK];
  const int w  = threadIdx.x >> 5;
  const int wm = w >> 2, wn = w & 3;
  const int mB = blockIdx.x * TM;
  const int nB = blockIdx.y * TN;
  const int l = threadIdx.x & 31, r = l & 15, hf = l >> 4;

  v8f acc[2][4];
#pragma unroll
  for (int i = 0; i < 2; ++i)
#pragma unroll
    for (int j = 0; j < 4; ++j) acc[i][j] = zero8();

  gemm_core(Ctx + (size_t)mB * E_, W + (size_t)nB * E_, E_, sA, sB, acc);

  const int m0 = mB + wm * 32;
  const int n0 = nB + wn * 64;
#pragma unroll
  for (int i = 0; i < 2; ++i)
#pragma unroll
    for (int j = 0; j < 4; ++j) {
      const int nb = n0 + 16 * j + r;
      const float bv = bias[nb];
#pragma unroll
      for (int t = 0; t < 8; ++t) {
        const int m = m0 + 16 * i + t + 8 * hf;
        out[(size_t)m * E_ + nb] = acc[i][j][t] + bv;
      }
    }
}

// ---------------------------------------------------------------------------
// Launch: cast(x, w_in, w_out) -> qkv GEMM -> flash attention -> out GEMM
// Workspace: Xbf(8M) Win(6M) Wout(2M) Q(8M) K(8M) Vt(8M) Ctx(8M) ~= 48 MB
// ---------------------------------------------------------------------------
extern "C" void kernel_launch(void* const* d_in, const int* in_sizes, int n_in,
                              void* d_out, int out_size, void* d_ws, size_t ws_size,
                              hipStream_t stream) {
  (void)in_sizes; (void)n_in; (void)out_size; (void)ws_size;
  const float* x      = (const float*)d_in[0];
  const float* w_in   = (const float*)d_in[1];
  const float* b_in   = (const float*)d_in[2];
  const float* w_out  = (const float*)d_in[3];
  const float* b_out  = (const float*)d_in[4];
  const int*   causal = (const int*)d_in[5];
  float* out = (float*)d_out;

  const size_t nX   = (size_t)B_ * S_ * E_;  // 4,194,304
  const size_t nWin = (size_t)3 * E_ * E_;   // 3,145,728
  const size_t nWo  = (size_t)E_ * E_;       // 1,048,576

  char* p = (char*)d_ws;
  bf16* Xbf   = (bf16*)p; p += nX * 2;
  bf16* Winb  = (bf16*)p; p += nWin * 2;
  bf16* Woutb = (bf16*)p; p += nWo * 2;
  bf16* Qb    = (bf16*)p; p += nX * 2;
  bf16* Kb    = (bf16*)p; p += nX * 2;
  bf16* Vtb   = (bf16*)p; p += nX * 2;
  bf16* Ctxb  = (bf16*)p; p += nX * 2;

  cast_f32_bf16<<<dim3((unsigned)((nX + 255) / 256)), 256, 0, stream>>>(x, Xbf, (int)nX);
  cast_f32_bf16<<<dim3((unsigned)((nWin + 255) / 256)), 256, 0, stream>>>(w_in, Winb, (int)nWin);
  cast_f32_bf16<<<dim3((unsigned)((nWo + 255) / 256)), 256, 0, stream>>>(w_out, Woutb, (int)nWo);

  // M=4096, N=3072: block tile 64M x 256N
  qkv_kernel<<<dim3(64, 12), 256, 0, stream>>>(Xbf, Winb, b_in, Qb, Kb, Vtb);

  // B*H*(S/128) blocks, 8 waves each (one 16-query tile per wave)
  attn_kernel<<<dim3(B_ * H_ * (S_ / 128)), 256, 0, stream>>>(Qb, Kb, Vtb, Ctxb, causal);

  // M=4096, N=1024: block tile 64M x 256N
  out_kernel<<<dim3(64, 4), 256, 0, stream>>>(Ctxb, Woutb, b_out, out);
}
